// LabelContrastiveLoss_87574383165821
// MI455X (gfx1250) — compile-verified, matrix operand order checked by reading, not run on hardware
//
#include <hip/hip_runtime.h>

typedef _Float16 half_t;
typedef __attribute__((ext_vector_type(16))) _Float16 v16h;
typedef __attribute__((ext_vector_type(8)))  _Float16 v8h;
typedef __attribute__((ext_vector_type(8)))  float    v8f;
typedef __attribute__((ext_vector_type(4)))  float    v4f;

#define DIM  256
#define NTAG 64
#define ROWS_PER_BLOCK 128   // 8 waves x 16 rows
#define INV_T 2.0f           // 1 / TEMPERATURE

// ---------------------------------------------------------------------------
// Kernel 0: normalize prototypes (eps 1e-8), store as f16 row-major [NTAG][DIM]
// ---------------------------------------------------------------------------
__global__ __launch_bounds__(64) void proto_norm_kernel(
    const float* __restrict__ emb, half_t* __restrict__ P) {
  const int t = threadIdx.x;            // one tag per thread, 64 threads
  const float* row = emb + (long)t * DIM;
  float ss = 0.f;
  for (int d = 0; d < DIM; d += 4) {
    v4f x = *(const v4f*)(row + d);
    ss += x[0]*x[0] + x[1]*x[1] + x[2]*x[2] + x[3]*x[3];
  }
  const float inv = 1.0f / fmaxf(sqrtf(ss), 1e-8f);
  half_t* prow = P + (long)t * DIM;
  for (int d = 0; d < DIM; d += 4) {
    v4f x = *(const v4f*)(row + d);
    prow[d + 0] = (half_t)(x[0] * inv);
    prow[d + 1] = (half_t)(x[1] * inv);
    prow[d + 2] = (half_t)(x[2] * inv);
    prow[d + 3] = (half_t)(x[3] * inv);
  }
}

// ---------------------------------------------------------------------------
// Kernel 1: main GEMM (f16 WMMA, fp32 accum) + per-row contrastive loss
// ---------------------------------------------------------------------------
__global__ __launch_bounds__(256) void loss_kernel(
    const float* __restrict__ feat, const int* __restrict__ labels,
    const half_t* __restrict__ proto, float* __restrict__ partials) {
  __shared__ half_t Pl[NTAG * DIM];     // 32 KB prototype tile
  __shared__ float  wred[8];

  const int tid = threadIdx.x;

  // cooperative copy proto -> LDS: 32768 B = 2048 x 16B, 256 threads x 8
  {
    const v8h* src = (const v8h*)proto;
    v8h*       dst = (v8h*)Pl;
#pragma unroll
    for (int i = 0; i < (NTAG * DIM / 8) / 256; ++i)
      dst[tid + i * 256] = src[tid + i * 256];
  }
  __syncthreads();

  const int wave   = tid >> 5;
  const int lane   = tid & 31;
  const int laneHi = lane >> 4;         // 0 or 1
  const int ln     = lane & 15;

  const long rowBase = (long)blockIdx.x * ROWS_PER_BLOCK + wave * 16;
  const float* frow  = feat + (rowBase + ln) * DIM;   // A row for this lane

  v8f acc[4];
#pragma unroll
  for (int nt = 0; nt < 4; ++nt)
#pragma unroll
    for (int i = 0; i < 8; ++i) acc[nt][i] = 0.f;

  float ss = 0.f;   // partial ||feature row||^2 (this lane's K subset)

#pragma unroll
  for (int ks = 0; ks < 8; ++ks) {
    const int kb = ks * 32;
    // A-fragment K layout (16-bit 16x32): lanes 0-15: K{kb..kb+7, kb+16..23},
    // lanes 16-31: K{kb+8..15, kb+24..31}
    const int c0 = kb + (laneHi ? 8 : 0);
    v4f x0 = *(const v4f*)(frow + c0);
    v4f x1 = *(const v4f*)(frow + c0 + 4);
    v4f x2 = *(const v4f*)(frow + c0 + 16);
    v4f x3 = *(const v4f*)(frow + c0 + 20);
    v16h a;
#pragma unroll
    for (int i = 0; i < 4; ++i) {
      a[i]      = (half_t)x0[i];
      a[i + 4]  = (half_t)x1[i];
      a[i + 8]  = (half_t)x2[i];
      a[i + 12] = (half_t)x3[i];
      ss += x0[i]*x0[i] + x1[i]*x1[i] + x2[i]*x2[i] + x3[i]*x3[i];
    }
#pragma unroll
    for (int nt = 0; nt < 4; ++nt) {
      // B-fragment: lane = column N (tag), same K striping as A
      const half_t* pr = Pl + (nt * 16 + ln) * DIM + c0;
      v8h blo = *(const v8h*)pr;          // ds_load_b128
      v8h bhi = *(const v8h*)(pr + 16);
      v16h b;
#pragma unroll
      for (int i = 0; i < 8; ++i) { b[i] = blo[i]; b[i + 8] = bhi[i]; }
      acc[nt] = __builtin_amdgcn_wmma_f32_16x16x32_f16(
          false, a, false, b, (short)0, acc[nt], false, false);
    }
  }

  // lanes (l, l+16) together cover one row's K range -> pair-sum the norms
  ss += __shfl_xor(ss, 16, 32);          // every lane: ||row(ln)||^2

  float lossAcc = 0.f;
#pragma unroll
  for (int r = 0; r < 8; ++r) {
    // C/D layout: VGPR r holds M=r (lanes 0-15) and M=r+8 (lanes 16-31)
    const int   rowIn = r + laneHi * 8;
    const float n2    = __shfl(ss, rowIn, 32);
    const float inv   = 1.0f / fmaxf(sqrtf(n2), 1e-12f);
    const int   lbl   = labels[rowBase + rowIn];
    float etot = 0.f, epos = 0.f;
#pragma unroll
    for (int nt = 0; nt < 4; ++nt) {
      const float sim = acc[nt][r] * inv;        // cosine similarity
      const float ex  = __expf(INV_T * sim);
      etot += ex;
      if ((nt * 16 + ln) == lbl) epos = ex;
    }
    // reduce across the 16 lanes of this half (xor 1,2,4,8 stays in-half)
#pragma unroll
    for (int m = 1; m <= 8; m <<= 1) {
      etot += __shfl_xor(etot, m, 32);
      epos += __shfl_xor(epos, m, 32);
    }
    if (ln == 0)
      lossAcc += __logf(etot - epos) - __logf(epos);   // log(neg) - log(pos)
  }
  lossAcc += __shfl_xor(lossAcc, 16, 32);
  if (lane == 0) wred[wave] = lossAcc;
  __syncthreads();
  if (tid == 0) {
    float s = 0.f;
#pragma unroll
    for (int w = 0; w < 8; ++w) s += wred[w];
    partials[blockIdx.x] = s;
  }
}

// ---------------------------------------------------------------------------
// Kernel 2: deterministic fixed-order reduction of per-block partials
// ---------------------------------------------------------------------------
__global__ __launch_bounds__(256) void reduce_kernel(
    const float* __restrict__ part, int n, float* __restrict__ out) {
  __shared__ float s[256];
  float v = 0.f;
  for (int i = threadIdx.x; i < n; i += 256) v += part[i];
  s[threadIdx.x] = v;
  __syncthreads();
  for (int st = 128; st > 0; st >>= 1) {
    if (threadIdx.x < st) s[threadIdx.x] += s[threadIdx.x + st];
    __syncthreads();
  }
  if (threadIdx.x == 0) out[0] = s[0];
}

// ---------------------------------------------------------------------------
extern "C" void kernel_launch(void* const* d_in, const int* in_sizes, int n_in,
                              void* d_out, int out_size, void* d_ws,
                              size_t ws_size, hipStream_t stream) {
  const float* features  = (const float*)d_in[0];   // [B,S,D] f32
  const int*   labels    = (const int*)d_in[1];     // [B,S] i32
  const float* label_emb = (const float*)d_in[2];   // [T,D] f32
  float*       out       = (float*)d_out;           // scalar f32

  half_t* P        = (half_t*)d_ws;                          // 32 KB
  float*  partials = (float*)((char*)d_ws + NTAG * DIM * 2); // per-block sums

  const int rows   = in_sizes[1];               // B*S = 131072
  const int blocks = rows / ROWS_PER_BLOCK;     // 1024

  proto_norm_kernel<<<1, 64, 0, stream>>>(label_emb, P);
  loss_kernel<<<blocks, 256, 0, stream>>>(features, labels, P, partials);
  reduce_kernel<<<1, 256, 0, stream>>>(partials, blocks, out);
}
